// ChebyASPIRELayer_80530636800166
// MI455X (gfx1250) — compile-verified, matrix operand order checked by reading, not run on hardware
//
#include <hip/hip_runtime.h>
#include <hip/hip_bf16.h>
#include <stdint.h>

// ---------------------------------------------------------------------------
// ChebyASPIRE layer for MI455X (gfx1250, wave32).
// Primary plan: densify X to bf16 once, each Gram apply = two
// v_wmma_f32_16x16x32_bf16 GEMMs (f32 accumulate). Tiling: 80(M) x 128(N)
// block tile, N is the fast grid dimension so sibling N-blocks reuse the same
// 3.2 MB A-strip out of L2 -> X streams from HBM ~once per GEMM:
// 40 GEMMs x 800 MB = 32 GB / 23.3 TB/s ~= 1.4 ms (HBM-bound).
// A-tiles are DMA'd into LDS with global_load_async_to_lds_b128 (ASYNCcnt).
// Fallback plan (small workspace): COO atomic SpMM with global_prefetch.
// ---------------------------------------------------------------------------

#define NUQ   20000      // n_users
#define NIQ   20000      // n_items
#define NNZQ  2000000
#define BQ    1024       // batch
#define DEGQ  20
#define MT    80         // M rows per block  (5 WMMA sub-tiles)
#define MSUB  5

typedef __attribute__((ext_vector_type(16))) __bf16          v16bf;
typedef __attribute__((ext_vector_type(8)))  float           v8f;
typedef __attribute__((ext_vector_type(8)))  unsigned short  us8;

static __device__ __forceinline__ unsigned short f32_to_bf16(float f) {
  union { float f; uint32_t u; } x; x.f = f;
  uint32_t r = x.u + 0x7FFFu + ((x.u >> 16) & 1u);   // round-to-nearest-even
  return (unsigned short)(r >> 16);
}

// ---------------------------------------------------------------- fill -----
__global__ void fill_u32_kernel(uint32_t* __restrict__ p, size_t n, uint32_t v) {
  size_t i = (size_t)blockIdx.x * blockDim.x + threadIdx.x;
  if (i < n) p[i] = v;
}

// ------------------------------------------------------- input transpose ---
// X_batch [B x NI] -> T0 [NI x B] (f32), optional bf16 copy for WMMA path.
__global__ void transpose_in_kernel(const float* __restrict__ xb,
                                    float* __restrict__ t0,
                                    unsigned short* __restrict__ t0b) {
  __shared__ float tile[32][33];
  int i0 = blockIdx.x * 32;           // item tile
  int b0 = blockIdx.y * 32;           // batch tile
  int tx = threadIdx.x, ty = threadIdx.y;   // block = (32, 8)
  for (int r = ty; r < 32; r += 8)
    tile[r][tx] = xb[(size_t)(b0 + r) * NIQ + (i0 + tx)];
  __syncthreads();
  for (int r = ty; r < 32; r += 8) {
    float v = tile[tx][r];            // = xb[b0+tx][i0+r]
    size_t o = (size_t)(i0 + r) * BQ + (b0 + tx);
    t0[o] = v;
    if (t0b) t0b[o] = f32_to_bf16(v);
  }
}

// W [NI x B] -> out [B x NI]
__global__ void transpose_out_kernel(const float* __restrict__ w,
                                     float* __restrict__ outp) {
  __shared__ float tile[32][33];
  int i0 = blockIdx.x * 32, b0 = blockIdx.y * 32;
  int tx = threadIdx.x, ty = threadIdx.y;
  for (int r = ty; r < 32; r += 8)
    tile[r][tx] = w[(size_t)(i0 + r) * BQ + (b0 + tx)];
  __syncthreads();
  for (int r = ty; r < 32; r += 8)
    outp[(size_t)(b0 + r) * NIQ + (i0 + tx)] = tile[tx][r];
}

// ---------------------------------------------------------- densify X -----
__global__ void densify_kernel(const int* __restrict__ rows,
                               const int* __restrict__ cols,
                               const float* __restrict__ vals,
                               unsigned short* __restrict__ xd) {
  int e = blockIdx.x * blockDim.x + threadIdx.x;
  if (e < NNZQ)
    xd[(size_t)rows[e] * NIQ + cols[e]] = f32_to_bf16(vals[e]);
}

// --------------------------------------------------------- WMMA GEMM ------
// C[M x B] = A * Bm. A bf16: row-major [M x K], or X^T via TRANS_A reading
// A[k*lda + m]. Bm bf16 row-major [K x B]. Block tile: MT(=80) x 128, 8
// waves; each wave owns MSUB(=5) 16x16 f32 accumulators; K-loop step 32.
// blockIdx.x = N tile (fast, for L2 reuse of the A strip), blockIdx.y = M.
template <bool TRANS_A, bool OUT_BF16>
__global__ void __launch_bounds__(256)
wmma_gemm_kernel(const unsigned short* __restrict__ A,
                 const unsigned short* __restrict__ Bm,
                 void* __restrict__ Cout, int K, int lda) {
  __shared__ __align__(16) unsigned short Bt[128 * 32];    // [n][k] transposed
  __shared__ __align__(16) unsigned short Araw[MT * 32];   // [m][k] row-major

  const int n0 = blockIdx.x * 128;
  const int m0 = blockIdx.y * MT;
  const int tid  = threadIdx.x;
  const int lane = tid & 31;
  const int wave = tid >> 5;             // 0..7 -> N sub-tile
  const int l16  = lane & 15;
  const int kh   = (lane < 16) ? 0 : 8;  // per-lane K-half select (ISA layout)

  v8f acc[MSUB];
#pragma unroll
  for (int s = 0; s < MSUB; ++s) acc[s] = (v8f){};

  for (int k0 = 0; k0 < K; k0 += 32) {
    // Stage B tile [32 x 128] -> Bt[n][k] (transposed) cooperatively.
    for (int c = tid; c < 512; c += 256) {
      int kr = c >> 4;                 // 0..31
      int nc = (c & 15) << 3;          // 0..120 step 8
      us8 d = *(const us8*)(Bm + (size_t)(k0 + kr) * BQ + n0 + nc);
#pragma unroll
      for (int j = 0; j < 8; ++j) Bt[(nc + j) * 32 + kr] = d[j];
    }
    if (TRANS_A) {
      // Stage A' tile: rows k0..k0+31 of X, cols m0..m0+MT-1 -> Araw[m][k].
      for (int c = tid; c < 32 * (MT / 8); c += 256) {   // 320 chunks of 8
        int kr = c / (MT / 8);
        int mc = (c % (MT / 8)) * 8;
        us8 d = *(const us8*)(A + (size_t)(k0 + kr) * lda + m0 + mc);
#pragma unroll
        for (int j = 0; j < 8; ++j) Araw[(mc + j) * 32 + kr] = d[j];
      }
    } else {
      // Stage A tile [MT x 32] row-major via async DMA to LDS (wave 0):
      // MT*4 = 320 x 16B chunks = 10 wave32 global_load_async_to_lds_b128.
      if (wave == 0) {
#pragma unroll
        for (int h = 0; h < MT / 8; ++h) {
          int chunk = h * 32 + lane;          // 0..319
          int row   = chunk >> 2;             // 0..MT-1
          int c8    = (chunk & 3) << 3;       // element offset 0,8,16,24
          uint32_t ldsb = (uint32_t)(uintptr_t)(Araw + row * 32 + c8);
          uint32_t goff = (uint32_t)(((size_t)(m0 + row) * lda + k0 + c8) * 2u);
          asm volatile("global_load_async_to_lds_b128 %0, %1, %2"
                       :: "v"(ldsb), "v"(goff), "s"(A)
                       : "memory");
        }
        asm volatile("s_wait_asynccnt 0x0" ::: "memory");
      }
    }
    __syncthreads();

    // Per-lane fragments (A: lane<16 holds K{0..7,16..23}, lane>=16 K{8..15,24..31}).
    union { us8 h[2]; v16bf v; } bfu;
    const int nloc = wave * 16 + l16;
    bfu.h[0] = *(const us8*)(Bt + nloc * 32 + kh);
    bfu.h[1] = *(const us8*)(Bt + nloc * 32 + 16 + kh);

#pragma unroll
    for (int s = 0; s < MSUB; ++s) {
      union { us8 h[2]; v16bf v; } af;
      const unsigned short* ap = Araw + (s * 16 + l16) * 32;
      af.h[0] = *(const us8*)(ap + kh);
      af.h[1] = *(const us8*)(ap + 16 + kh);
      acc[s] = __builtin_amdgcn_wmma_f32_16x16x32_bf16(
          /*neg_a=*/false, af.v, /*neg_b=*/false, bfu.v,
          /*c_mod=*/(short)0, acc[s], /*reuse_a=*/false, /*reuse_b=*/true);
    }

    // Prefetch next K-slab of B into the cache hierarchy.
    if (k0 + 32 < K)
      __builtin_prefetch(Bm + (size_t)(k0 + 32 + (tid >> 4)) * BQ + n0 + ((tid & 15) << 3), 0, 0);

    __syncthreads();
  }

  // C/D layout: lane<16 -> M = v, lane>=16 -> M = v+8; N = lane&15.
  const int n = n0 + wave * 16 + l16;
#pragma unroll
  for (int s = 0; s < MSUB; ++s) {
#pragma unroll
    for (int v = 0; v < 8; ++v) {
      int m = m0 + s * 16 + v + ((lane < 16) ? 0 : 8);
      if constexpr (OUT_BF16)
        ((unsigned short*)Cout)[(size_t)m * BQ + n] = f32_to_bf16(acc[s][v]);
      else
        ((float*)Cout)[(size_t)m * BQ + n] = acc[s][v];
    }
  }
}

// ----------------------------------------------- fallback: atomic SpMM ----
// out[sidx[e], :] += vals[e] * in[gidx[e], :]
__global__ void spmm_atomic_kernel(const int* __restrict__ gidx,
                                   const int* __restrict__ sidx,
                                   const float* __restrict__ vals,
                                   const float* __restrict__ tin,
                                   float* __restrict__ outp) {
  int col = blockIdx.y * 256 + threadIdx.x;
  int e0 = blockIdx.x * 64;
  int e1 = e0 + 64;
  for (int e = e0; e < e1; ++e) {
    int g = gidx[e], s = sidx[e];
    float v = vals[e];
    if (e + 1 < e1)
      __builtin_prefetch(tin + (size_t)gidx[e + 1] * BQ + col, 0, 0);  // global_prefetch_b8
    atomicAdd(outp + (size_t)s * BQ + col, v * tin[(size_t)g * BQ + col]);
  }
}

// ------------------------------------------------- Chebyshev updates ------
__global__ void update_first_kernel(const float* __restrict__ U,
                                    const float* __restrict__ T0,
                                    float* __restrict__ Tc, float* __restrict__ W,
                                    unsigned short* __restrict__ Tcb,
                                    const float* __restrict__ coeffs,
                                    const float* __restrict__ pmid,
                                    const float* __restrict__ phalf) {
  size_t i = (size_t)blockIdx.x * blockDim.x + threadIdx.x;
  float mid = pmid[0], half = phalf[0];
  float c0 = coeffs[0], c1 = coeffs[1];
  float t0 = T0[i];
  float t1 = (U[i] - mid * t0) / half;
  Tc[i] = t1;
  W[i]  = c0 * t0 + c1 * t1;
  if (Tcb) Tcb[i] = f32_to_bf16(t1);
}

__global__ void update_step_kernel(const float* __restrict__ U,
                                   const float* __restrict__ Tc,
                                   float* __restrict__ TpToTn, float* __restrict__ W,
                                   unsigned short* __restrict__ Tcb,
                                   const float* __restrict__ coeffs, int k,
                                   const float* __restrict__ pmid,
                                   const float* __restrict__ phalf) {
  size_t i = (size_t)blockIdx.x * blockDim.x + threadIdx.x;
  float mid = pmid[0], half = phalf[0], c = coeffs[k];
  float tn = 2.0f * (U[i] - mid * Tc[i]) / half - TpToTn[i];
  TpToTn[i] = tn;                 // becomes the new T_curr after pointer swap
  W[i] += c * tn;
  if (Tcb) Tcb[i] = f32_to_bf16(tn);
}

// --------------------------------------------------------------- driver ---
extern "C" void kernel_launch(void* const* d_in, const int* in_sizes, int n_in,
                              void* d_out, int out_size, void* d_ws, size_t ws_size,
                              hipStream_t stream) {
  (void)in_sizes; (void)n_in; (void)out_size;
  const float* Xbatch = (const float*)d_in[0];
  const int*   rows   = (const int*)d_in[1];
  const int*   cols   = (const int*)d_in[2];
  const float* vals   = (const float*)d_in[3];
  const float* coeffs = (const float*)d_in[4];
  const float* pmid   = (const float*)d_in[5];
  const float* phalf  = (const float*)d_in[6];
  float* out = (float*)d_out;

  const size_t NIB = (size_t)NIQ * BQ;     // 20.48M elems
  char* p = (char*)d_ws;
  auto take = [&](size_t bytes) -> char* {
    char* q = p; p += (bytes + 255) & ~(size_t)255; return q;
  };
  float* Tp = (float*)take(NIB * 4);       // T_prev (starts as T0)
  float* Tc = (float*)take(NIB * 4);       // T_curr
  float* U  = (float*)take(NIB * 4);       // Gram output
  float* W  = (float*)take(NIB * 4);       // accumulator
  size_t used_common = (size_t)(p - (char*)d_ws);

  const size_t needDense = used_common + NIB * 2              // Tcb
                         + (size_t)NUQ * BQ * 2               // Yb
                         + (size_t)NUQ * NIQ * 2 + 4096;      // X dense bf16
  const bool dense = (ws_size >= needDense);

  unsigned short *Tcb = nullptr, *Yb = nullptr, *XD = nullptr;
  float* Yf = nullptr;
  if (dense) {
    Tcb = (unsigned short*)take(NIB * 2);
    Yb  = (unsigned short*)take((size_t)NUQ * BQ * 2);
    XD  = (unsigned short*)take((size_t)NUQ * NIQ * 2);
  } else {
    Yf = (float*)take((size_t)NUQ * BQ * 4);
  }

  // T0 = X_batch^T (+ bf16 copy for the WMMA path)
  transpose_in_kernel<<<dim3(NIQ / 32, BQ / 32), dim3(32, 8), 0, stream>>>(Xbatch, Tp, Tcb);

  if (dense) {
    const size_t words = (size_t)NUQ * NIQ / 2;   // bf16 pairs
    fill_u32_kernel<<<(unsigned)((words + 255) / 256), 256, 0, stream>>>((uint32_t*)XD, words, 0u);
    densify_kernel<<<(NNZQ + 255) / 256, 256, 0, stream>>>(rows, cols, vals, XD);
  }

  for (int k = 1; k <= DEGQ; ++k) {
    const float* Tin = (k == 1) ? Tp : Tc;
    if (dense) {
      // Y = X @ T   (bf16 in, f32 acc, bf16 out)
      wmma_gemm_kernel<false, true><<<dim3(BQ / 128, NUQ / MT), 256, 0, stream>>>(
          XD, Tcb, (void*)Yb, NIQ, NIQ);
      // U = X^T @ Y (bf16 in, f32 out)
      wmma_gemm_kernel<true, false><<<dim3(BQ / 128, NIQ / MT), 256, 0, stream>>>(
          XD, Yb, (void*)U, NUQ, NIQ);
    } else {
      fill_u32_kernel<<<(unsigned)(((size_t)NUQ * BQ + 255) / 256), 256, 0, stream>>>(
          (uint32_t*)Yf, (size_t)NUQ * BQ, 0u);
      spmm_atomic_kernel<<<dim3(NNZQ / 64, BQ / 256), 256, 0, stream>>>(cols, rows, vals, Tin, Yf);
      fill_u32_kernel<<<(unsigned)((NIB + 255) / 256), 256, 0, stream>>>(
          (uint32_t*)U, NIB, 0u);
      spmm_atomic_kernel<<<dim3(NNZQ / 64, BQ / 256), 256, 0, stream>>>(rows, cols, vals, Yf, U);
    }
    if (k == 1) {
      update_first_kernel<<<(unsigned)(NIB / 256), 256, 0, stream>>>(
          U, Tp, Tc, W, Tcb, coeffs, pmid, phalf);
    } else {
      update_step_kernel<<<(unsigned)(NIB / 256), 256, 0, stream>>>(
          U, Tc, Tp, W, Tcb, coeffs, k, pmid, phalf);
      float* t = Tp; Tp = Tc; Tc = t;   // (T_prev, T_curr) <- (T_curr, T_next)
    }
  }

  // out = W^T  [B x NI]
  transpose_out_kernel<<<dim3(NIQ / 32, BQ / 32), dim3(32, 8), 0, stream>>>(W, out);
}
